// GCNModel_53412213293437
// MI455X (gfx1250) — matrix operand order repaired and checked by reading, not compiled, not run on hardware
//
#include <hip/hip_runtime.h>
#include <hip/hip_bf16.h>
#include <stdint.h>

typedef __attribute__((ext_vector_type(16))) _Float16 v16h;
typedef __attribute__((ext_vector_type(8)))  float    v8f;

// ---------------------------------------------------------------------------
// Utility kernels
// ---------------------------------------------------------------------------
__global__ void fill_f32(float* __restrict__ p, float v, int n) {
  int i = blockIdx.x * blockDim.x + threadIdx.x;
  if (i < n) p[i] = v;
}

__global__ void deg_accum(const int* __restrict__ col, const float* __restrict__ w,
                          float* __restrict__ deg, int nE) {
  int e = blockIdx.x * blockDim.x + threadIdx.x;
  if (e < nE) atomicAdd(deg + col[e], w[e]);
}

__global__ void deg_to_dinv(float* __restrict__ d, int n) {
  int i = blockIdx.x * blockDim.x + threadIdx.x;
  if (i < n) {
    float v = d[i];
    d[i] = (v > 0.f) ? rsqrtf(fmaxf(v, 1e-12f)) : 0.f;
  }
}

__global__ void edge_norm_k(const int* __restrict__ row, const int* __restrict__ col,
                            const float* __restrict__ w, const float* __restrict__ dinv,
                            float* __restrict__ nrm, int nE) {
  int e = blockIdx.x * blockDim.x + threadIdx.x;
  if (e < nE) nrm[e] = dinv[row[e]] * w[e] * dinv[col[e]];
}

// per (edge, 4-feature group): hout[col] += norm_e * xw[row]
template <int F>
__global__ void scatter_k(const int* __restrict__ row, const int* __restrict__ col,
                          const float* __restrict__ nrm, const float* __restrict__ xw,
                          float* __restrict__ hout, int total) {
  constexpr int GRP = F >> 2;           // float4 groups per edge (2/4/8)
  int t = blockIdx.x * blockDim.x + threadIdx.x;
  if (t >= total) return;
  int e = t / GRP;                      // power of 2 -> shift
  int g = (t - e * GRP) << 2;           // feature offset
  int r = row[e];
  int c = col[e];
  float nm = nrm[e];
  const float4 v = *(const float4*)(xw + (size_t)r * F + g);
  float* dst = hout + (size_t)c * F + g;
  atomicAdd(dst + 0, nm * v.x);
  atomicAdd(dst + 1, nm * v.y);
  atomicAdd(dst + 2, nm * v.z);
  atomicAdd(dst + 3, nm * v.w);
}

// pooling: per (node, feature)
__global__ void pool_accum_k(const float* __restrict__ h, const int* __restrict__ batch,
                             float* __restrict__ pool, float* __restrict__ cnt, int nNodes) {
  int t = blockIdx.x * blockDim.x + threadIdx.x;
  if (t >= nNodes * 32) return;
  int i = t >> 5;
  int f = t & 31;
  int b = batch[i];
  atomicAdd(pool + (size_t)b * 32 + f, h[(size_t)i * 32 + f]);
  if (f == 0) atomicAdd(cnt + b, 1.0f);
}

__global__ void pool_div_k(float* __restrict__ pool, const float* __restrict__ cnt, int total) {
  int t = blockIdx.x * blockDim.x + threadIdx.x;
  if (t >= total) return;
  int g = t >> 5;
  pool[t] = pool[t] / fmaxf(cnt[g], 1.0f);
}

// ---------------------------------------------------------------------------
// WMMA GEMM with fused epilogue.
//   Xw[M,Fout]   = act(Hin)[M,Fin] @ W[Fin,Fout]          (act = relu if RELU_IN)
//   Hout[M,Fout] = bias + dinv^2 * Xw                     (if FUSE_INIT)
//   Xw[M,Fout]  += bias                                   (if !FUSE_INIT, bias!=null)
// One wave32 per (16-row tile, 16-col tile); K zero-padded to 32.
// Requires nRows % 16 == 0 (500000 and 4096 both qualify).
//
// B is staged in LDS pre-shuffled into WMMA register layout: each lane's 16
// f16 elements are contiguous (32B) -> per-wave B fetch = 2x ds_load_b128.
// A loads: float4 (b128) for Fin in {8,16,32}; scalar clamp+cndmask for
// Fin==9 (36B rows, unaligned). No divergent load guards anywhere.
//
// Layouts per ISA 7.12.2 (wave32):
//   A (16x32 f16): half=lane>>4, M=lane&15; elem e -> K = 16*(e>>3)+8*half+(e&7)
//   B (32x16 f16): N=lane&15;              elem e -> K = 16*(e>>3)+8*half+(e&7)
//   D (16x16 f32): VGPR r -> M = r + 8*half, N = lane&15
// ---------------------------------------------------------------------------
template <int Fin, int Fout, bool RELU_IN, bool FUSE_INIT>
__global__ void gemm_wmma_k(const float* __restrict__ Hin, const float* __restrict__ W,
                            const float* __restrict__ bias, float* __restrict__ Xw,
                            float* __restrict__ Hout, const float* __restrict__ dinv,
                            int nRows) {
  constexpr int CT = (Fout + 15) >> 4;      // column tiles
  __shared__ _Float16 sB[CT * 32 * 16];     // [ct][lane][e], 32B per lane, <= 2KB

  // cooperative staging: sB[ct*512 + lane*16 + e] = Wpad[16*(e>>3)+8*(lane>>4)+(e&7)][ct*16+(lane&15)]
  for (int t = threadIdx.x; t < CT * 32 * 16; t += blockDim.x) {
    int e    = t & 15;
    int ln   = (t >> 4) & 31;
    int ct   = t >> 9;
    int kb   = ((e >> 3) << 4) + ((ln >> 4) << 3) + (e & 7);
    int c    = ct * 16 + (ln & 15);
    float v  = (kb < Fin && c < Fout) ? W[kb * Fout + c] : 0.f;
    sB[t] = (_Float16)v;
  }
  __syncthreads();

  int wave = (blockIdx.x * blockDim.x + threadIdx.x) >> 5;
  int lane = threadIdx.x & 31;
  int nodeTile = wave / CT;
  int colTile  = wave - nodeTile * CT;
  int m0 = nodeTile * 16;
  if (m0 >= nRows) return;

  int half = lane >> 4;
  int m    = lane & 15;
  int gcol = colTile * 16 + (lane & 15);

  const float* Arow = Hin + (m0 + m) * Fin;

  auto cv = [](float v) -> _Float16 {
    if constexpr (RELU_IN) v = fmaxf(v, 0.f);
    return (_Float16)v;
  };

  v16h a;
  if constexpr (Fin == 32) {
    const float4* A0 = (const float4*)(Arow + (half << 3));        // K=8h+0..7
    const float4* A1 = (const float4*)(Arow + 16 + (half << 3));   // K=16+8h+0..7
    float4 q0 = A0[0], q1 = A0[1], q2 = A1[0], q3 = A1[1];
    a[0] = cv(q0.x); a[1] = cv(q0.y); a[2]  = cv(q0.z); a[3]  = cv(q0.w);
    a[4] = cv(q1.x); a[5] = cv(q1.y); a[6]  = cv(q1.z); a[7]  = cv(q1.w);
    a[8] = cv(q2.x); a[9] = cv(q2.y); a[10] = cv(q2.z); a[11] = cv(q2.w);
    a[12] = cv(q3.x); a[13] = cv(q3.y); a[14] = cv(q3.z); a[15] = cv(q3.w);
  } else if constexpr (Fin == 16) {
    const float4* A0 = (const float4*)(Arow + (half << 3));        // K=8h+0..7
    float4 q0 = A0[0], q1 = A0[1];
    a[0] = cv(q0.x); a[1] = cv(q0.y); a[2] = cv(q0.z); a[3] = cv(q0.w);
    a[4] = cv(q1.x); a[5] = cv(q1.y); a[6] = cv(q1.z); a[7] = cv(q1.w);
#pragma unroll
    for (int e = 8; e < 16; ++e) a[e] = (_Float16)0.f;             // K>=16
  } else if constexpr (Fin == 8) {
    const float4* A0 = (const float4*)Arow;                        // K=0..7
    float4 q0 = A0[0], q1 = A0[1];
    bool lo = (half == 0);                                         // half=1 -> K>=8
    a[0] = lo ? cv(q0.x) : (_Float16)0.f; a[1] = lo ? cv(q0.y) : (_Float16)0.f;
    a[2] = lo ? cv(q0.z) : (_Float16)0.f; a[3] = lo ? cv(q0.w) : (_Float16)0.f;
    a[4] = lo ? cv(q1.x) : (_Float16)0.f; a[5] = lo ? cv(q1.y) : (_Float16)0.f;
    a[6] = lo ? cv(q1.z) : (_Float16)0.f; a[7] = lo ? cv(q1.w) : (_Float16)0.f;
#pragma unroll
    for (int e = 8; e < 16; ++e) a[e] = (_Float16)0.f;
  } else {
    // generic (Fin==9): clamped unconditional loads + branchless blend
#pragma unroll
    for (int e = 0; e < 16; ++e) {
      int ka = ((e >> 3) << 4) + (half << 3) + (e & 7);
      int kc = (ka < Fin) ? ka : (Fin - 1);
      float tv = Arow[kc];
      a[e] = (ka < Fin) ? cv(tv) : (_Float16)0.f;
    }
  }

  // B fragment: 32 contiguous bytes per lane -> 2x ds_load_b128
  v16h b = *((const v16h*)sB + (colTile << 5) + lane);

  v8f c = {};
  c = __builtin_amdgcn_wmma_f32_16x16x32_f16(false, a, false, b, (short)0, c, false, false);

  if (gcol < Fout) {                         // single exec region
    int row0 = m0 + (half << 3);
    int base = row0 * Fout + gcol;
    if constexpr (FUSE_INIT) {
      float bv = bias[gcol];
#pragma unroll
      for (int r = 0; r < 8; ++r) {
        float d = dinv[row0 + r];
        Xw[base + r * Fout]   = c[r];
        Hout[base + r * Fout] = bv + d * d * c[r];
      }
    } else {
      float bv = (bias != nullptr) ? bias[gcol] : 0.f;
#pragma unroll
      for (int r = 0; r < 8; ++r)
        Xw[base + r * Fout] = c[r] + bv;
    }
  }
}

// ---------------------------------------------------------------------------
static inline int blocks_for(long long threads, int bs) {
  return (int)((threads + bs - 1) / bs);
}

extern "C" void kernel_launch(void* const* d_in, const int* in_sizes, int n_in,
                              void* d_out, int out_size, void* d_ws, size_t ws_size,
                              hipStream_t stream) {
  (void)n_in; (void)ws_size;
  const float* x    = (const float*)d_in[0];
  const int*   ei   = (const int*)d_in[1];   // [2, E]
  const float* ew   = (const float*)d_in[2];
  const int*   bat  = (const int*)d_in[3];
  const float* W1   = (const float*)d_in[4];
  const float* b1   = (const float*)d_in[5];
  const float* W2   = (const float*)d_in[6];
  const float* b2   = (const float*)d_in[7];
  const float* W3   = (const float*)d_in[8];
  const float* b3   = (const float*)d_in[9];
  const float* Wl   = (const float*)d_in[10];
  const float* bl   = (const float*)d_in[11];
  float* out        = (float*)d_out;

  const int N = in_sizes[0] / 9;      // 500000 (multiple of 16)
  const int E = in_sizes[1] / 2;      // 8000000
  const int G = out_size / 2;         // 4096   (multiple of 16)

  const int* row = ei;
  const int* col = ei + E;

  // workspace layout (floats); offsets 16B-aligned for these N,E
  float* ws   = (float*)d_ws;
  float* dinv = ws;                           // N
  float* nrm  = dinv + (size_t)N;             // E
  float* X    = nrm + (size_t)E;              // N*32 (gemm output, scatter source)
  float* H0   = X   + (size_t)N * 32;         // N*32
  float* H1   = H0  + (size_t)N * 32;         // N*32
  float* pool = H1  + (size_t)N * 32;         // G*32
  float* cnt  = pool + (size_t)G * 32;        // G (contiguous with pool)

  const int BS = 256;
  const int nodeTiles = N / 16;

  // ---- normalization: deg (init 1.0 = self-loop) -> dinv -> per-edge norms ----
  fill_f32<<<blocks_for(N, BS), BS, 0, stream>>>(dinv, 1.0f, N);
  deg_accum<<<blocks_for(E, BS), BS, 0, stream>>>(col, ew, dinv, E);
  deg_to_dinv<<<blocks_for(N, BS), BS, 0, stream>>>(dinv, N);
  edge_norm_k<<<blocks_for(E, BS), BS, 0, stream>>>(row, col, ew, dinv, nrm, E);

  // ---- layer 1: x[N,9] -> H0[N,8]  (gemm fuses self-loop+bias seed) ----
  gemm_wmma_k<9, 8, false, true><<<blocks_for((long long)nodeTiles * 32, BS), BS, 0, stream>>>(
      x, W1, b1, X, H0, dinv, N);
  scatter_k<8><<<blocks_for((long long)E * 2, BS), BS, 0, stream>>>(row, col, nrm, X, H0, E * 2);

  // ---- layer 2: relu(H0)[N,8] -> H1[N,16] ----
  gemm_wmma_k<8, 16, true, true><<<blocks_for((long long)nodeTiles * 32, BS), BS, 0, stream>>>(
      H0, W2, b2, X, H1, dinv, N);
  scatter_k<16><<<blocks_for((long long)E * 4, BS), BS, 0, stream>>>(row, col, nrm, X, H1, E * 4);

  // ---- layer 3: relu(H1)[N,16] -> H0[N,32] (no output relu) ----
  gemm_wmma_k<16, 32, true, true><<<blocks_for((long long)nodeTiles * 2 * 32, BS), BS, 0, stream>>>(
      H1, W3, b3, X, H0, dinv, N);
  scatter_k<32><<<blocks_for((long long)E * 8, BS), BS, 0, stream>>>(row, col, nrm, X, H0, E * 8);

  // ---- global mean pool over G graphs ----
  fill_f32<<<blocks_for(G * 32 + G, BS), BS, 0, stream>>>(pool, 0.0f, G * 32 + G); // pool+cnt
  pool_accum_k<<<blocks_for((long long)N * 32, BS), BS, 0, stream>>>(H0, bat, pool, cnt, N);
  pool_div_k<<<blocks_for(G * 32, BS), BS, 0, stream>>>(pool, cnt, G * 32);

  // ---- final linear: out[G,2] = pooled[G,32] @ Wl[32,2] + bl, via WMMA ----
  gemm_wmma_k<32, 2, false, false><<<blocks_for((long long)(G / 16) * 32, BS), BS, 0, stream>>>(
      pool, Wl, bl, out, nullptr, nullptr, G);
}
